// BertInfiniSelfAttention_80650895884563
// MI455X (gfx1250) — matrix operand order repaired and checked by reading, not compiled, not run on hardware
//
#include <hip/hip_runtime.h>

// ---------------------------------------------------------------------------
// Problem constants (reference: B=2, S=768, H=768, NH=12, D=64)
// ---------------------------------------------------------------------------
#define BB   2
#define SS   768
#define HH   768
#define NHH  12
#define DD   64
#define KCROWS (NHH * SS)       // 9216
#define CHUNK  64               // keys per attention inner-loop iteration
#define NIT    (KCROWS / CHUNK) // 144

typedef __attribute__((ext_vector_type(16))) __bf16 v16bf;
typedef __attribute__((ext_vector_type(8)))  float  v8f;

union Frag16 { v16bf v; uint4 q[2]; };

__device__ __forceinline__ unsigned short f2bf(float f) {
  unsigned u = __float_as_uint(f);
  u += 0x7FFFu + ((u >> 16) & 1u);          // round-to-nearest-even
  return (unsigned short)(u >> 16);
}

__device__ __forceinline__ v8f wmma_bf16(const Frag16 &a, const Frag16 &b, v8f c) {
  // emits v_wmma_f32_16x16x32_bf16
  return __builtin_amdgcn_wmma_f32_16x16x32_bf16(
      /*neg_a=*/false, a.v, /*neg_b=*/false, b.v,
      /*c_mod=*/(short)0, c, /*reuse_a=*/false, /*reuse_b=*/false);
}

// ---------------------------------------------------------------------------
// Async global->LDS copy (CDNA5 GLOBAL_LOAD_ASYNC_TO_LDS_B128, ASYNCcnt).
// ISA 00_overview §10.2: flat LDS address low 32 bits == LDS byte offset;
// flat global address == global address. AS-qualified pointers are built via
// integer casts (implicitly convertible to generic if the builtin wants that).
// If the builtin is unavailable, emit the instruction via inline asm
// (CDNA5 ISA opcode 98, GV addressing mode).
// ---------------------------------------------------------------------------
#if defined(__has_builtin)
#if __has_builtin(__builtin_amdgcn_load_to_lds)
#define HAVE_ASYNC_LDS_BUILTIN 1
#endif
#if __has_builtin(__builtin_amdgcn_s_wait_asynccnt)
#define HAVE_WAIT_ASYNC_BUILTIN 1
#endif
#endif

typedef __attribute__((address_space(3))) void lds_void;
typedef __attribute__((address_space(1))) void glb_void;

__device__ __forceinline__ void cp16_to_lds(const void* g, void* l) {
#ifdef HAVE_ASYNC_LDS_BUILTIN
  __builtin_amdgcn_load_to_lds(
      (glb_void*)(unsigned long long)g,
      (lds_void*)(unsigned)(unsigned long long)l,
      /*size=*/16u, /*offset=*/0, /*cpol=*/0u);
#else
  const unsigned           ldsOff = (unsigned)(unsigned long long)l;
  const unsigned long long gaddr  = (unsigned long long)g;
  asm volatile("global_load_async_to_lds_b128 %0, %1, off"
               :: "v"(ldsOff), "v"(gaddr) : "memory");
#endif
}

__device__ __forceinline__ void wait_async_lds() {
#ifdef HAVE_WAIT_ASYNC_BUILTIN
  __builtin_amdgcn_s_wait_asynccnt(0);
#else
  asm volatile("s_wait_asynccnt 0" ::: "memory");
#endif
}

// ---------------------------------------------------------------------------
// Generic tiled GEMM:  C(bf16) = A(fp32) @ B(fp32 or bf16) [+ bias]
//   grid = (M/64, N/64, Z), block = 128 (4 waves)
//   outMode 0: scatter to [B,NH,S,D] bf16 (QKV head split)
//   outMode 1: contiguous rows*64 per z block (memory-compressed K/V)
// ---------------------------------------------------------------------------
template <bool B_IS_BF16>
__global__ __launch_bounds__(128)
void wmma_gemm_kernel(const float* __restrict__ A, int lda, int aZmod, long aZstride,
                      const void* __restrict__ Braw, int ldb, long bZstride,
                      const float* __restrict__ bias,
                      unsigned short* __restrict__ Cout, long cZstride, int outMode,
                      int K)
{
  __shared__ __align__(16) unsigned short aT[64 * 32];   // [m][k]
  __shared__ __align__(16) unsigned short bT[64 * 32];   // [n][k] (transposed)

  const int z    = blockIdx.z;
  const int tid  = threadIdx.x;
  const int wave = tid >> 5;
  const int lane = tid & 31;
  const int half = lane >> 4;
  const int l15  = lane & 15;

  const float* Ab = A + (long)(z % aZmod) * aZstride + (long)blockIdx.x * 64 * lda;

  v8f acc[4] = {};

  for (int kk = 0; kk < K; kk += 32) {
    __syncthreads();
    // ---- stage A: 64 rows x 32 k, fp32 -> bf16, vectorized float4 loads ----
    for (int i = tid; i < 512; i += 128) {          // 512 quads of 4 floats
      int row = i >> 3;                             // 8 quads per row
      int c4  = (i & 7) * 4;
      const float4 f = *(const float4*)(Ab + (long)row * lda + kk + c4);
      unsigned short* dst = &aT[row * 32 + c4];
      dst[0] = f2bf(f.x); dst[1] = f2bf(f.y); dst[2] = f2bf(f.z); dst[3] = f2bf(f.w);
    }
    // ---- stage B transposed: B[k][n] -> bT[n][k] ----
    if (B_IS_BF16) {
      const unsigned short* Bb = (const unsigned short*)Braw + (long)z * bZstride
                               + (long)blockIdx.y * 64;
      for (int i = tid; i < 256; i += 128) {        // 256 chunks of 8 bf16
        int krow = i >> 3;
        int c8   = (i & 7) * 8;
        uint4 u = *(const uint4*)(Bb + (long)(kk + krow) * ldb + c8);
        const unsigned short* p = (const unsigned short*)&u;
        #pragma unroll
        for (int j = 0; j < 8; ++j) bT[(c8 + j) * 32 + krow] = p[j];
      }
    } else {
      const float* Bb = (const float*)Braw + (long)z * bZstride + (long)blockIdx.y * 64;
      for (int i = tid; i < 512; i += 128) {        // 512 quads of 4 floats
        int krow = i >> 4;                          // 16 quads per k-row
        int c4   = (i & 15) * 4;
        const float4 f = *(const float4*)(Bb + (long)(kk + krow) * ldb + c4);
        bT[(c4 + 0) * 32 + krow] = f2bf(f.x);
        bT[(c4 + 1) * 32 + krow] = f2bf(f.y);
        bT[(c4 + 2) * 32 + krow] = f2bf(f.z);
        bT[(c4 + 3) * 32 + krow] = f2bf(f.w);
      }
    }
    __syncthreads();

    // ---- fragments + 4 WMMAs (one 16-row strip per wave, 4 N-subtiles) ----
    Frag16 afr;
    const int arow = wave * 16 + l15;
    afr.q[0] = *(const uint4*)&aT[arow * 32 + half * 8];        // K = half*8..+7
    afr.q[1] = *(const uint4*)&aT[arow * 32 + 16 + half * 8];   // K = 16+half*8..+7
    #pragma unroll
    for (int t = 0; t < 4; ++t) {
      Frag16 bfr;
      const int n = t * 16 + l15;
      bfr.q[0] = *(const uint4*)&bT[n * 32 + half * 8];         // K = half*8..+7
      bfr.q[1] = *(const uint4*)&bT[n * 32 + 16 + half * 8];    // K = 16+half*8..+7
      acc[t] = wmma_bf16(afr, bfr, acc[t]);
    }
  }

  // ---- epilogue: bias add, bf16 convert, scatter ----
  #pragma unroll
  for (int t = 0; t < 4; ++t) {
    #pragma unroll
    for (int r = 0; r < 8; ++r) {
      const int m = blockIdx.x * 64 + wave * 16 + half * 8 + r;
      const int n = blockIdx.y * 64 + t * 16 + l15;
      float v = acc[t][r];
      if (bias) v += bias[n];
      long off;
      if (outMode == 0) {   // [b*S+s, h*64+d] -> [B,NH,S,D]
        const int b = m / SS, s = m % SS, h = n >> 6, d = n & 63;
        off = (((long)(b * NHH + h) * SS + s) << 6) + d;
      } else {              // contiguous [rows x 64] per z
        off = (long)z * cZstride + (long)m * 64 + n;
      }
      Cout[off] = f2bf(v);
    }
  }
}

// ---------------------------------------------------------------------------
// Flash attention over the 9216-wide compressed-memory key axis.
//   grid = (S/64, NH, B), block = 128 (4 waves; each wave owns 16 q rows)
//   Double-buffered pipeline: async kc copy (ASYNCcnt) + vc global loads for
//   chunk i+1 are issued BEFORE computing chunk i; the vc transpose-commit to
//   LDS happens after compute so both copies retire under 16 WMMAs + softmax.
// ---------------------------------------------------------------------------
__global__ __launch_bounds__(128)
void attn_kernel(const unsigned short* __restrict__ q16,
                 const unsigned short* __restrict__ kc16,
                 const unsigned short* __restrict__ vc16,
                 const float* __restrict__ mask,
                 const float* __restrict__ gate,
                 float* __restrict__ out)
{
  __shared__ __align__(16) unsigned short kcL[2][CHUNK * 64];   // [key][d]  16KB
  __shared__ __align__(16) unsigned short vcL[2][64 * CHUNK];   // [d][key]  16KB
  __shared__ __align__(16) unsigned short pL[4][16 * CHUNK];    // per-wave   8KB

  const int b    = blockIdx.z;
  const int h    = blockIdx.y;
  const int qt   = blockIdx.x;
  const int tid  = threadIdx.x;
  const int wave = tid >> 5;
  const int lane = tid & 31;
  const int half = lane >> 4;
  const int l15  = lane & 15;

  // ---- Q A-fragments (K=64 split into two K=32 groups), kept in registers ----
  Frag16 qa[2];
  {
    const int row = qt * 64 + wave * 16 + l15;
    const unsigned short* qp = q16 + (((long)(b * NHH + h) * SS + row) << 6);
    #pragma unroll
    for (int g = 0; g < 2; ++g) {
      qa[g].q[0] = *(const uint4*)(qp + g * 32 + half * 8);
      qa[g].q[1] = *(const uint4*)(qp + g * 32 + 16 + half * 8);
    }
  }

  v8f o[4] = {};
  float rmax[8], rsum[8];
  #pragma unroll
  for (int r = 0; r < 8; ++r) { rmax[r] = -1e30f; rsum[r] = 0.0f; }

  const unsigned short* kcB = kc16 + (long)b * KCROWS * DD;
  const unsigned short* vcB = vc16 + (long)b * KCROWS * DD;
  const float* maskB = mask + (long)b * KCROWS;

  // ---- staging helpers (wave-uniform control flow) ----
  auto issue_kc = [&](int c, int buf) {            // contiguous 8KB, async
    const char* src = (const char*)(kcB + (long)c * DD);
    char*       dst = (char*)&kcL[buf][0];
    #pragma unroll
    for (int j = 0; j < 4; ++j) {
      const int off = (tid + j * 128) * 16;
      cp16_to_lds(src + off, dst + off);
    }
  };
  auto load_vc = [&](int c, uint4* vr) {           // global -> regs (LOADcnt)
    #pragma unroll
    for (int j = 0; j < 4; ++j) {
      const int i = tid + j * 128;
      const int key = i >> 3, c8 = (i & 7) * 8;
      vr[j] = *(const uint4*)(vcB + (long)(c + key) * DD + c8);
    }
  };
  auto commit_vc = [&](int buf, const uint4* vr) { // regs -> LDS transposed
    #pragma unroll
    for (int j = 0; j < 4; ++j) {
      const int i = tid + j * 128;
      const int key = i >> 3, c8 = (i & 7) * 8;
      const unsigned short* p = (const unsigned short*)&vr[j];
      #pragma unroll
      for (int jj = 0; jj < 8; ++jj) vcL[buf][(c8 + jj) * CHUNK + key] = p[jj];
    }
  };

  // ---- pipeline prologue: stage chunk 0 into buffer 0 ----
  uint4 vreg[4];
  issue_kc(0, 0);
  load_vc(0, vreg);
  commit_vc(0, vreg);
  wait_async_lds();
  __syncthreads();

  for (int it = 0; it < NIT; ++it) {
    const int  cur  = it & 1;
    const int  nxt  = cur ^ 1;
    const int  c    = it * CHUNK;
    const bool more = (it + 1) < NIT;

    // issue next chunk's copies before computing on the current one
    if (more) { issue_kc(c + CHUNK, nxt); load_vc(c + CHUNK, vreg); }

    // ---- scores: 4 key groups x (K=64 -> 2 WMMAs) = 8 WMMAs ----
    v8f s[4] = {};
    #pragma unroll
    for (int kg = 0; kg < 4; ++kg) {
      #pragma unroll
      for (int g = 0; g < 2; ++g) {
        Frag16 bf;
        const unsigned short* base =
            &kcL[cur][(kg * 16 + l15) * 64 + g * 32 + half * 8];
        bf.q[0] = *(const uint4*)base;
        bf.q[1] = *(const uint4*)(base + 16);
        s[kg] = wmma_bf16(qa[g], bf, s[kg]);
      }
    }

    float mval[4];
    #pragma unroll
    for (int kg = 0; kg < 4; ++kg) mval[kg] = maskB[c + kg * 16 + l15];

    // ---- online softmax: one reduction tree covers all 64 keys ----
    float corr[8];
    #pragma unroll
    for (int r = 0; r < 8; ++r) {
      float a0 = s[0][r] * 0.125f + mval[0];       // 1/sqrt(64)
      float a1 = s[1][r] * 0.125f + mval[1];
      float a2 = s[2][r] * 0.125f + mval[2];
      float a3 = s[3][r] * 0.125f + mval[3];
      float mx = fmaxf(fmaxf(a0, a1), fmaxf(a2, a3));
      #pragma unroll
      for (int d = 1; d < 16; d <<= 1) mx = fmaxf(mx, __shfl_xor(mx, d, 32));
      mx = fmaxf(mx, rmax[r]);
      float p0 = __expf(a0 - mx);
      float p1 = __expf(a1 - mx);
      float p2 = __expf(a2 - mx);
      float p3 = __expf(a3 - mx);
      float ps = (p0 + p1) + (p2 + p3);
      #pragma unroll
      for (int d = 1; d < 16; d <<= 1) ps += __shfl_xor(ps, d, 32);
      float cr = __expf(rmax[r] - mx);
      rsum[r] = rsum[r] * cr + ps;
      rmax[r] = mx;
      corr[r] = cr;
      unsigned short* pp = &pL[wave][(half * 8 + r) * CHUNK + l15];
      pp[0]  = f2bf(p0);
      pp[16] = f2bf(p1);
      pp[32] = f2bf(p2);
      pp[48] = f2bf(p3);
    }
    #pragma unroll
    for (int t = 0; t < 4; ++t)
      #pragma unroll
      for (int r = 0; r < 8; ++r) o[t][r] *= corr[r];

    // ---- context: O += P[16x64] @ vc[64x64] -> 8 WMMAs ----
    Frag16 pa[2];
    #pragma unroll
    for (int kb = 0; kb < 2; ++kb) {
      pa[kb].q[0] = *(const uint4*)&pL[wave][l15 * CHUNK + kb * 32 + half * 8];
      pa[kb].q[1] = *(const uint4*)&pL[wave][l15 * CHUNK + kb * 32 + 16 + half * 8];
    }
    #pragma unroll
    for (int t = 0; t < 4; ++t) {
      #pragma unroll
      for (int kb = 0; kb < 2; ++kb) {
        Frag16 bv;
        const unsigned short* base =
            &vcL[cur][(t * 16 + l15) * CHUNK + kb * 32 + half * 8];
        bv.q[0] = *(const uint4*)base;
        bv.q[1] = *(const uint4*)(base + 16);
        o[t] = wmma_bf16(pa[kb], bv, o[t]);
      }
    }

    // commit next chunk's vc transpose after compute (loads retired by now)
    if (more) commit_vc(nxt, vreg);
    wait_async_lds();        // ASYNCcnt not covered by barrier-implied waits
    __syncthreads();
  }

  // ---- epilogue: normalize, sigmoid gate, write fp32 [B,S,NH,D] ----
  const float g = 1.0f / (1.0f + __expf(-gate[h]));
  #pragma unroll
  for (int t = 0; t < 4; ++t) {
    #pragma unroll
    for (int r = 0; r < 8; ++r) {
      const int row = half * 8 + r;
      const int s   = qt * 64 + wave * 16 + row;
      const int d   = t * 16 + l15;
      out[(((long)(b * SS + s) * NHH + h) << 6) + d] = o[t][r] / rsum[r] * g;
    }
  }
}

// ---------------------------------------------------------------------------
// Launcher
// inputs: 0 hidden_states, 1 attention_mask, 2 Wq, 3 bq, 4 Wk, 5 bk,
//         6 Wv, 7 bv, 8 gate, 9 mem_keys, 10 mem_values
// ---------------------------------------------------------------------------
extern "C" void kernel_launch(void* const* d_in, const int* in_sizes, int n_in,
                              void* d_out, int out_size, void* d_ws, size_t ws_size,
                              hipStream_t stream) {
  const float* hs   = (const float*)d_in[0];
  const float* mask = (const float*)d_in[1];
  const float* Wq   = (const float*)d_in[2];
  const float* bq   = (const float*)d_in[3];
  const float* Wk   = (const float*)d_in[4];
  const float* bk   = (const float*)d_in[5];
  const float* Wv   = (const float*)d_in[6];
  const float* bv   = (const float*)d_in[7];
  const float* gate = (const float*)d_in[8];
  const float* mk   = (const float*)d_in[9];
  const float* mv   = (const float*)d_in[10];
  float* out = (float*)d_out;

  const long QKV_E = (long)BB * NHH * SS * DD;     // 1,179,648 elements each
  unsigned short* q16  = (unsigned short*)d_ws;
  unsigned short* k16  = q16  + QKV_E;
  unsigned short* v16  = k16  + QKV_E;
  unsigned short* kc16 = v16  + QKV_E;
  unsigned short* vc16 = kc16 + QKV_E;

  dim3 blk(128);

  // 1) QKV projections: [1536x768] @ [768x768] + bias -> bf16 [B,NH,S,D]
  dim3 gQKV(BB * SS / 64, HH / 64, 1);
  wmma_gemm_kernel<false><<<gQKV, blk, 0, stream>>>(hs, HH, 1, 0L, Wq, HH, 0L, bq,
                                                    q16, 0L, 0, HH);
  wmma_gemm_kernel<false><<<gQKV, blk, 0, stream>>>(hs, HH, 1, 0L, Wk, HH, 0L, bk,
                                                    k16, 0L, 0, HH);
  wmma_gemm_kernel<false><<<gQKV, blk, 0, stream>>>(hs, HH, 1, 0L, Wv, HH, 0L, bv,
                                                    v16, 0L, 0, HH);

  // 2) Memory compression per (b,h): mem[h][768x768] @ {k,v}[b,h][768x64]
  dim3 gMEM(SS / 64, 1, BB * NHH);
  wmma_gemm_kernel<true><<<gMEM, blk, 0, stream>>>(mk, HH, NHH, (long)HH * HH,
                                                   k16, DD, (long)SS * DD, nullptr,
                                                   kc16, (long)SS * DD, 1, HH);
  wmma_gemm_kernel<true><<<gMEM, blk, 0, stream>>>(mv, HH, NHH, (long)HH * HH,
                                                   v16, DD, (long)SS * DD, nullptr,
                                                   vc16, (long)SS * DD, 1, HH);

  // 3) Flash attention over 9216-wide key axis + sigmoid gate
  dim3 gATT(SS / 64, NHH, BB);
  attn_kernel<<<gATT, blk, 0, stream>>>(q16, kc16, vc16, mask, gate, out);
}
